// Invariance_Fea_Extractor_71554155151863
// MI455X (gfx1250) — compile-verified, matrix operand order checked by reading, not run on hardware
//
#include <hip/hip_runtime.h>

// ---------------------------------------------------------------------------
// Problem constants (match reference)
// ---------------------------------------------------------------------------
constexpr int   Nn    = 25000;
constexpr int   Ee    = 400000;
constexpr int   ATOM  = 64;
constexpr int   INV   = 64;
constexpr int   DNODE = 128;
constexpr int   DEDGE = 128;
constexpr int   HID   = 384;   // 6*INV
constexpr int   F1    = 384;   // feat dim of inv block (3*INV + 3*ATOM)
constexpr int   F3    = 320;   // feat dim of mrg block (2*DNODE + INV)

constexpr int   TILES = Ee / 16;        // 25000 edge tiles
constexpr int   KS1   = F1 / 32;        // 12
constexpr int   NT1   = HID / 16;       // 24
constexpr int   KS2   = HID / 32;       // 12
constexpr int   NT2   = DNODE / 16;     // 8
constexpr int   KS3   = F3 / 32;        // 10
constexpr int   NT3   = 128 / 16;       // 8  (mrg hidden = 2*INV = 128)
constexpr int   KS4   = 128 / 32;       // 4
constexpr int   NT4   = DEDGE / 16;     // 8

// Workspace layout (bytes)
constexpr size_t OFF_MASK   = 0;                              // E * 4
constexpr size_t OFF_LAST   = OFF_MASK + (size_t)Ee * 4;      // 256 pad
constexpr size_t OFF_B1     = OFF_LAST + 256;                 // 24*12*512 bf16
constexpr size_t OFF_B2     = OFF_B1 + (size_t)NT1 * KS1 * 512 * 2;
constexpr size_t OFF_B3     = OFF_B2 + (size_t)NT2 * KS2 * 512 * 2;
constexpr size_t OFF_B4     = OFF_B3 + (size_t)NT3 * KS3 * 512 * 2;
constexpr size_t OFF_B1P    = OFF_B4 + (size_t)NT4 * KS4 * 512 * 2;
constexpr size_t OFF_NODEBF = OFF_B1P + 2048;                 // N*128 bf16

// ---------------------------------------------------------------------------
// WMMA types / helpers
// ---------------------------------------------------------------------------
typedef __attribute__((ext_vector_type(16))) __bf16 v16bf;
typedef __attribute__((ext_vector_type(8)))  float  v8f;

union BF16x16 { uint4 q[2]; v16bf v; };

__device__ __forceinline__ unsigned short f2bf(float f) {
  union { float f; unsigned u; } x; x.f = f;
  unsigned r = (x.u + 0x7fffu + ((x.u >> 16) & 1u)) >> 16;   // RNE
  return (unsigned short)r;
}

__device__ __forceinline__ void st_bf4(unsigned short* p, float a, float b,
                                       float c, float d) {
  uint2 u;
  u.x = (unsigned)f2bf(a) | ((unsigned)f2bf(b) << 16);
  u.y = (unsigned)f2bf(c) | ((unsigned)f2bf(d) << 16);
  *reinterpret_cast<uint2*>(p) = u;   // ds_store_b64
}

// A-fragment (16x32 bf16) from a row-major LDS tile.
// Layout (ISA 7.12.2): lanes 0-15 row M=lane, K pairs {k0..k0+7, k0+16..k0+23};
// lanes 16-31 row M=lane-16, K {k0+8..k0+15, k0+24..k0+31}.
__device__ __forceinline__ v16bf load_frag_lds(const unsigned short* base,
                                               int stride, int lane, int k0) {
  int half = (lane >> 4) & 1;
  int row  = lane & 15;
  const unsigned short* p = base + row * stride + k0 + half * 8;
  BF16x16 u;
  u.q[0] = *reinterpret_cast<const uint4*>(p);
  u.q[1] = *reinterpret_cast<const uint4*>(p + 16);
  return u.v;
}

// B-fragment from fragment-major packed weights (16 contiguous bf16 per lane).
__device__ __forceinline__ v16bf load_frag_glb(const unsigned short* p) {
  BF16x16 u;
  u.q[0] = *reinterpret_cast<const uint4*>(p);
  u.q[1] = *reinterpret_cast<const uint4*>(p + 8);
  return u.v;
}

__device__ __forceinline__ v8f wmma_bf16(v16bf a, v16bf b, v8f c) {
  return __builtin_amdgcn_wmma_f32_16x16x32_bf16(false, a, false, b,
                                                 (short)0, c, false, false);
}

// Fragment coordinate for packing: mirrors load_frag_lds mapping (B = W^T tile).
__device__ __forceinline__ void frag_coord(int lane, int i, int k, int n,
                                           int& Kg, int& Ng) {
  Kg = k * 32 + ((lane >> 4) & 1) * 8 + (i & 7) + (i >> 3) * 16;
  Ng = n * 16 + (lane & 15);
}

__device__ __forceinline__ float silu(float x) {
  return x / (1.0f + __expf(-x));
}

// ---------------------------------------------------------------------------
// Software-pipelined tile GEMM: C[16,NT*16] += A[16,KS*32] * B
//   - B fragments stream from global (L2-resident packed weights) through a
//     depth-8 rotating buffer so loads run ~8 WMMAs ahead of use.
//   - next k-step's A fragment is fetched from LDS during the current burst.
// ---------------------------------------------------------------------------
template <int KS, int NT>
__device__ __forceinline__ void run_gemm(const unsigned short* fa, int stride,
                                         const unsigned short* B, int lane,
                                         v8f* acc) {
  constexpr int T = KS * NT;
  constexpr int D = 8;                     // prefetch distance (fragments)
  const unsigned short* Bl = B + lane * 16;

  v16bf bb[D];
#pragma unroll
  for (int t = 0; t < D; ++t) {
    int k = t / NT, n = t % NT;
    bb[t] = load_frag_glb(Bl + (n * KS + k) * 512);
  }
  v16bf a_cur = load_frag_lds(fa, stride, lane, 0);
  v16bf a_next = a_cur;
#pragma unroll
  for (int t = 0; t < T; ++t) {
    int k = t / NT, n = t % NT;
    if (n == 0 && k + 1 < KS)
      a_next = load_frag_lds(fa, stride, lane, (k + 1) * 32);
    v16bf bc = bb[t % D];
    if (t + D < T) {
      int t2 = t + D, k2 = t2 / NT, n2 = t2 % NT;
      bb[t % D] = load_frag_glb(Bl + (n2 * KS + k2) * 512);
    }
    acc[n] = wmma_bf16(a_cur, bc, acc[n]);
    if (n == NT - 1) a_cur = a_next;
  }
}

// ---------------------------------------------------------------------------
// Prep kernels
// ---------------------------------------------------------------------------
__global__ __launch_bounds__(256) void zero_f32_k(float* p, int n) {
  int i = blockIdx.x * 256 + threadIdx.x;
  if (i < n) p[i] = 0.0f;
}

__global__ void init_last_k(int* last) { last[0] = -1; }

__global__ __launch_bounds__(256) void mask1_k(const int* __restrict__ SRC,
                                               const int* __restrict__ DST,
                                               int* __restrict__ mask,
                                               int* __restrict__ last) {
  int e = blockIdx.x * 256 + threadIdx.x;
  if (e >= Ee) return;
  int sv = SRC[e], dv = DST[e];
  int ss = (e + 1 < Ee) ? SRC[e + 1] : 0;
  int ds = (e + 1 < Ee) ? DST[e + 1] : 0;
  int m = (sv == ss) && (sv != dv) && (ss != ds);
  mask[e] = m;
  if (m) atomicMax(last, e);
}

__global__ void mask2_k(int* mask, const int* last) {
  int l = last[0];
  if (l >= 0) mask[l] = 0;   // torch nonzero(mask)[:-1]
}

// Pack inv_w1 (with folded angle_w2) into B1 fragments + folded bias b1p.
__global__ __launch_bounds__(256) void pack_w1_k(const float* __restrict__ W1,
                                                 const float* __restrict__ AW2,
                                                 unsigned short* __restrict__ B1) {
  int idx = blockIdx.x * 256 + threadIdx.x;
  if (idx >= NT1 * KS1 * 512) return;
  int i = idx & 15, lane = (idx >> 4) & 31;
  int kn = idx >> 9, k = kn % KS1, n = kn / KS1;
  int Kg, Ng; frag_coord(lane, i, k, n, Kg, Ng);
  float v;
  if (Kg < 320) {
    v = W1[Ng * F1 + Kg];
  } else {
    int jj = Kg - 320;
    float s = 0.f;
    for (int kk = 0; kk < 64; ++kk)
      s += W1[Ng * F1 + 320 + kk] * AW2[kk * 64 + jj];
    v = s;
  }
  B1[idx] = f2bf(v);
}

__global__ __launch_bounds__(256) void pack_b1p_k(const float* __restrict__ W1,
                                                  const float* __restrict__ ib1,
                                                  const float* __restrict__ AB2,
                                                  float* __restrict__ b1p) {
  int n = blockIdx.x * 256 + threadIdx.x;
  if (n >= HID) return;
  float s = ib1[n];
  for (int kk = 0; kk < 64; ++kk)
    s += W1[n * F1 + 320 + kk] * AB2[kk];
  b1p[n] = s;
}

__global__ __launch_bounds__(256) void pack_gen_k(const float* __restrict__ W,
                                                  unsigned short* __restrict__ B,
                                                  int ntiles, int ksteps, int din) {
  int idx = blockIdx.x * 256 + threadIdx.x;
  if (idx >= ntiles * ksteps * 512) return;
  int i = idx & 15, lane = (idx >> 4) & 31;
  int kn = idx >> 9, k = kn % ksteps, n = kn / ksteps;
  int Kg, Ng; frag_coord(lane, i, k, n, Kg, Ng);
  float v = (Kg < din) ? W[Ng * din + Kg] : 0.0f;
  B[idx] = f2bf(v);
}

__global__ __launch_bounds__(256) void node_to_bf_k(const float* __restrict__ node,
                                                    unsigned short* __restrict__ nb) {
  int i = blockIdx.x * 256 + threadIdx.x;
  if (i < Nn * DNODE) nb[i] = f2bf(node[i]);
}

// ---------------------------------------------------------------------------
// GEMM 1: per 16-edge tile -> inv block -> atomic segment-sum into node
// ---------------------------------------------------------------------------
__global__ __launch_bounds__(128)
void gemm_node_k(const float* __restrict__ atom, const float* __restrict__ el,
                 const float* __restrict__ evec, const int* __restrict__ SRC,
                 const int* __restrict__ DST, const int* __restrict__ mask,
                 const float* __restrict__ aw1, const float* __restrict__ ab1,
                 const float* __restrict__ ag,  const float* __restrict__ abn,
                 const float* __restrict__ b1p, const float* __restrict__ ib1,
                 const float* __restrict__ ig,  const float* __restrict__ ibn,
                 const float* __restrict__ ib2,
                 const unsigned short* __restrict__ B1,
                 const unsigned short* __restrict__ B2,
                 float* __restrict__ node) {
  __shared__ unsigned short sA[4][16 * F1];   // 12 KB per wave
  const int wave = threadIdx.x >> 5;
  const int lane = threadIdx.x & 31;
  const int tile = blockIdx.x * 4 + wave;
  unsigned short* fa = sA[wave];
  const long e0 = (long)tile * 16;

  // ---- gather feat tile (bf16) : [a_src | a_dst | el | a_dst_s | el_s | h] ----
  {
    int r = lane >> 1, half = lane & 1;
    long e = e0 + r;
    if (half == 0) {
      int sv = SRC[e], dv = DST[e];
      const float* ps = atom + (long)sv * ATOM;
      const float* pd = atom + (long)dv * ATOM;
      const float* pe = el + e * 64;
      if (e0 + 16 + r < Ee) __builtin_prefetch(el + (e0 + 16 + r) * 64, 0, 1);
#pragma unroll
      for (int j = 0; j < 64; j += 4) {
        float4 a = *reinterpret_cast<const float4*>(ps + j);
        float4 b = *reinterpret_cast<const float4*>(pd + j);
        float4 c = *reinterpret_cast<const float4*>(pe + j);
        st_bf4(fa + r * F1 + j,        a.x, a.y, a.z, a.w);
        st_bf4(fa + r * F1 + 64 + j,   b.x, b.y, b.z, b.w);
        st_bf4(fa + r * F1 + 128 + j,  c.x, c.y, c.z, c.w);
      }
    } else {
      bool has = (e + 1 < Ee);
      int d2 = has ? DST[e + 1] : 0;         // shift zero-pads -> index 0 row
      const float* pd2 = atom + (long)d2 * ATOM;
#pragma unroll
      for (int j = 0; j < 64; j += 4) {
        float4 a = *reinterpret_cast<const float4*>(pd2 + j);
        st_bf4(fa + r * F1 + 192 + j, a.x, a.y, a.z, a.w);
        if (has) {
          float4 b = *reinterpret_cast<const float4*>(el + (e + 1) * 64 + j);
          st_bf4(fa + r * F1 + 256 + j, b.x, b.y, b.z, b.w);
        } else {
          st_bf4(fa + r * F1 + 256 + j, 0.f, 0.f, 0.f, 0.f);
        }
      }
      // cos(angle) between masked unit edge vectors e and e+1
      float cosv = 0.0f;
      int m0 = mask[e];
      int m1 = has ? mask[e + 1] : 0;
      if (m0 && m1) {
        float ax = evec[e * 3], ay = evec[e * 3 + 1], az = evec[e * 3 + 2];
        float bx = evec[(e + 1) * 3], by = evec[(e + 1) * 3 + 1],
              bz = evec[(e + 1) * 3 + 2];
        float ra = rsqrtf(ax * ax + ay * ay + az * az);
        float rb = rsqrtf(bx * bx + by * by + bz * bz);
        cosv = (ax * bx + ay * by + az * bz) * ra * rb;
      }
      // angle hidden: h = silu(ln(cos*w1 + b1))   (w2 folded into B1)
      float h[64]; float s1 = 0.f;
#pragma unroll
      for (int j = 0; j < 64; ++j) { h[j] = cosv * aw1[j] + ab1[j]; s1 += h[j]; }
      float mean = s1 * (1.0f / 64.0f), s2 = 0.f;
#pragma unroll
      for (int j = 0; j < 64; ++j) { float d = h[j] - mean; s2 += d * d; }
      float rstd = rsqrtf(s2 * (1.0f / 64.0f) + 1e-6f);
#pragma unroll
      for (int j = 0; j < 64; j += 4) {
        float x0 = silu((h[j]     - mean) * rstd * ag[j]     + abn[j]);
        float x1 = silu((h[j + 1] - mean) * rstd * ag[j + 1] + abn[j + 1]);
        float x2 = silu((h[j + 2] - mean) * rstd * ag[j + 2] + abn[j + 2]);
        float x3 = silu((h[j + 3] - mean) * rstd * ag[j + 3] + abn[j + 3]);
        st_bf4(fa + r * F1 + 320 + j, x0, x1, x2, x3);
      }
    }
  }
  __syncthreads();

  // ---- GEMM1: [16,384] x [384,384], 24 register-resident C tiles, pipelined ----
  v8f acc[NT1];
#pragma unroll
  for (int n = 0; n < NT1; ++n) acc[n] = v8f{};
  run_gemm<KS1, NT1>(fa, F1, B1, lane, acc);

  // ---- epilogue: bias, mask fix, LayerNorm(384), SiLU -> bf16 A2 tile ----
  const int l16 = lane & 15;
  const int mbase = (lane >> 4) * 8;     // C layout: lanes>=16 hold rows 8..15
  int mrow[8]; int drow[8];
#pragma unroll
  for (int j = 0; j < 8; ++j) {
    mrow[j] = mask[e0 + mbase + j];
    drow[j] = DST[e0 + mbase + j];
  }
  float s1[8], s2[8];
#pragma unroll
  for (int j = 0; j < 8; ++j) { s1[j] = 0.f; s2[j] = 0.f; }
#pragma unroll
  for (int n = 0; n < NT1; ++n) {
    int c = n * 16 + l16;
    float bp = b1p[c], ib = ib1[c];
#pragma unroll
    for (int j = 0; j < 8; ++j) {
      float x = acc[n][j] + bp;
      if (!mrow[j]) x = ib;              // feat row zeroed -> pre-LN = inv_b1
      acc[n][j] = x;
      s1[j] += x; s2[j] += x * x;
    }
  }
#pragma unroll
  for (int off = 1; off < 16; off <<= 1) {
#pragma unroll
    for (int j = 0; j < 8; ++j) {
      s1[j] += __shfl_xor(s1[j], off, 32);
      s2[j] += __shfl_xor(s2[j], off, 32);
    }
  }
  float mean[8], rstd[8];
#pragma unroll
  for (int j = 0; j < 8; ++j) {
    mean[j] = s1[j] * (1.0f / HID);
    float var = s2[j] * (1.0f / HID) - mean[j] * mean[j];
    rstd[j] = rsqrtf(var + 1e-6f);
  }
  __syncthreads();
#pragma unroll
  for (int n = 0; n < NT1; ++n) {
    int c = n * 16 + l16;
    float g = ig[c], bb = ibn[c];
#pragma unroll
    for (int j = 0; j < 8; ++j) {
      float x = silu((acc[n][j] - mean[j]) * rstd[j] * g + bb);
      fa[(mbase + j) * F1 + c] = f2bf(x);
    }
  }
  __syncthreads();

  // ---- GEMM2: [16,384] x [384,128] -> atomic add into node[dst] ----
  v8f a2[NT2];
#pragma unroll
  for (int n = 0; n < NT2; ++n) a2[n] = v8f{};
  run_gemm<KS2, NT2>(fa, F1, B2, lane, a2);
#pragma unroll
  for (int n = 0; n < NT2; ++n) {
    int c = n * 16 + l16;
    float bo = ib2[c];
#pragma unroll
    for (int j = 0; j < 8; ++j) {
      unsafeAtomicAdd(&node[(long)drow[j] * DNODE + c], a2[n][j] + bo);
    }
  }
}

// ---------------------------------------------------------------------------
// GEMM 3: per 16-edge tile -> mrg block -> edge output
// ---------------------------------------------------------------------------
__global__ __launch_bounds__(128)
void gemm_edge_k(const unsigned short* __restrict__ nodebf,
                 const float* __restrict__ el,
                 const int* __restrict__ SRC, const int* __restrict__ DST,
                 const float* __restrict__ mb1, const float* __restrict__ mg,
                 const float* __restrict__ mbn, const float* __restrict__ mb2,
                 const unsigned short* __restrict__ B3,
                 const unsigned short* __restrict__ B4,
                 float* __restrict__ edge_out) {
  __shared__ unsigned short sB[4][16 * F3];   // 10 KB per wave
  const int wave = threadIdx.x >> 5;
  const int lane = threadIdx.x & 31;
  const int tile = blockIdx.x * 4 + wave;
  unsigned short* fb = sB[wave];
  const long e0 = (long)tile * 16;

  // ---- gather [node_src | node_dst | el] (bf16) ----
  {
    int r = lane >> 1, half = lane & 1;
    long e = e0 + r;
    int sv = SRC[e], dv = DST[e];
    const unsigned short* ns = nodebf + (long)sv * DNODE;
    const unsigned short* nd = nodebf + (long)dv * DNODE;
    const float* pe = el + e * 64;
#pragma unroll
    for (int c4 = 0; c4 < 160; c4 += 4) {
      int c = half * 160 + c4;
      if (c < 128) {
        *reinterpret_cast<uint2*>(fb + r * F3 + c) =
            *reinterpret_cast<const uint2*>(ns + c);
      } else if (c < 256) {
        *reinterpret_cast<uint2*>(fb + r * F3 + c) =
            *reinterpret_cast<const uint2*>(nd + (c - 128));
      } else {
        float4 t = *reinterpret_cast<const float4*>(pe + (c - 256));
        st_bf4(fb + r * F3 + c, t.x, t.y, t.z, t.w);
      }
    }
  }
  __syncthreads();

  // ---- GEMM: [16,320] x [320,128], pipelined ----
  v8f acc[NT3];
#pragma unroll
  for (int n = 0; n < NT3; ++n) acc[n] = v8f{};
  run_gemm<KS3, NT3>(fb, F3, B3, lane, acc);

  // ---- bias + LayerNorm(128) + SiLU -> bf16 A2 ----
  const int l16 = lane & 15;
  const int mbase = (lane >> 4) * 8;
  float s1[8], s2[8];
#pragma unroll
  for (int j = 0; j < 8; ++j) { s1[j] = 0.f; s2[j] = 0.f; }
#pragma unroll
  for (int n = 0; n < NT3; ++n) {
    int c = n * 16 + l16;
    float bp = mb1[c];
#pragma unroll
    for (int j = 0; j < 8; ++j) {
      float x = acc[n][j] + bp;
      acc[n][j] = x;
      s1[j] += x; s2[j] += x * x;
    }
  }
#pragma unroll
  for (int off = 1; off < 16; off <<= 1) {
#pragma unroll
    for (int j = 0; j < 8; ++j) {
      s1[j] += __shfl_xor(s1[j], off, 32);
      s2[j] += __shfl_xor(s2[j], off, 32);
    }
  }
  float mean[8], rstd[8];
#pragma unroll
  for (int j = 0; j < 8; ++j) {
    mean[j] = s1[j] * (1.0f / 128.0f);
    float var = s2[j] * (1.0f / 128.0f) - mean[j] * mean[j];
    rstd[j] = rsqrtf(var + 1e-6f);
  }
  __syncthreads();
#pragma unroll
  for (int n = 0; n < NT3; ++n) {
    int c = n * 16 + l16;
    float g = mg[c], bb = mbn[c];
#pragma unroll
    for (int j = 0; j < 8; ++j) {
      float x = silu((acc[n][j] - mean[j]) * rstd[j] * g + bb);
      fb[(mbase + j) * 128 + c] = f2bf(x);   // reuse LDS, stride 128
    }
  }
  __syncthreads();

  // ---- GEMM: [16,128] x [128,128] -> edge output ----
  v8f a2[NT4];
#pragma unroll
  for (int n = 0; n < NT4; ++n) a2[n] = v8f{};
  run_gemm<KS4, NT4>(fb, 128, B4, lane, a2);
#pragma unroll
  for (int n = 0; n < NT4; ++n) {
    int c = n * 16 + l16;
    float bo = mb2[c];
#pragma unroll
    for (int j = 0; j < 8; ++j) {
      edge_out[(e0 + mbase + j) * DEDGE + c] = a2[n][j] + bo;
    }
  }
}

// ---------------------------------------------------------------------------
// Launch
// ---------------------------------------------------------------------------
extern "C" void kernel_launch(void* const* d_in, const int* in_sizes, int n_in,
                              void* d_out, int out_size, void* d_ws, size_t ws_size,
                              hipStream_t stream) {
  const float* atom = (const float*)d_in[0];
  const float* el   = (const float*)d_in[1];
  const float* evec = (const float*)d_in[2];
  const int*   eidx = (const int*)d_in[3];
  const int*   SRC  = eidx;
  const int*   DST  = eidx + Ee;
  const float* aw1  = (const float*)d_in[4];
  const float* ab1  = (const float*)d_in[5];
  const float* ag   = (const float*)d_in[6];
  const float* abn  = (const float*)d_in[7];
  const float* aw2  = (const float*)d_in[8];
  const float* ab2  = (const float*)d_in[9];
  const float* iw1  = (const float*)d_in[10];
  const float* ib1  = (const float*)d_in[11];
  const float* ig   = (const float*)d_in[12];
  const float* ibn  = (const float*)d_in[13];
  const float* iw2  = (const float*)d_in[14];
  const float* ib2  = (const float*)d_in[15];
  const float* mw1  = (const float*)d_in[16];
  const float* mb1  = (const float*)d_in[17];
  const float* mg   = (const float*)d_in[18];
  const float* mbn  = (const float*)d_in[19];
  const float* mw2  = (const float*)d_in[20];
  const float* mb2  = (const float*)d_in[21];

  char* ws = (char*)d_ws;
  int*            mask   = (int*)(ws + OFF_MASK);
  int*            last   = (int*)(ws + OFF_LAST);
  unsigned short* B1     = (unsigned short*)(ws + OFF_B1);
  unsigned short* B2     = (unsigned short*)(ws + OFF_B2);
  unsigned short* B3     = (unsigned short*)(ws + OFF_B3);
  unsigned short* B4     = (unsigned short*)(ws + OFF_B4);
  float*          b1p    = (float*)(ws + OFF_B1P);
  unsigned short* nodebf = (unsigned short*)(ws + OFF_NODEBF);

  float* node = (float*)d_out;              // [N,128]
  float* edge = node + (long)Nn * DNODE;    // [E,128]

  // 1. zero node accumulator, init last-index
  zero_f32_k<<<(Nn * DNODE + 255) / 256, 256, 0, stream>>>(node, Nn * DNODE);
  init_last_k<<<1, 1, 0, stream>>>(last);

  // 2. mask (with "drop last true index")
  mask1_k<<<(Ee + 255) / 256, 256, 0, stream>>>(SRC, DST, mask, last);
  mask2_k<<<1, 1, 0, stream>>>(mask, last);

  // 3. pack weights into WMMA B-fragment layout (bf16)
  pack_w1_k<<<(NT1 * KS1 * 512 + 255) / 256, 256, 0, stream>>>(iw1, aw2, B1);
  pack_b1p_k<<<(HID + 255) / 256, 256, 0, stream>>>(iw1, ib1, ab2, b1p);
  pack_gen_k<<<(NT2 * KS2 * 512 + 255) / 256, 256, 0, stream>>>(iw2, B2, NT2, KS2, HID);
  pack_gen_k<<<(NT3 * KS3 * 512 + 255) / 256, 256, 0, stream>>>(mw1, B3, NT3, KS3, F3);
  pack_gen_k<<<(NT4 * KS4 * 512 + 255) / 256, 256, 0, stream>>>(mw2, B4, NT4, KS4, 128);

  // 4. inv block + segment_sum (atomics)
  gemm_node_k<<<TILES / 4, 128, 0, stream>>>(atom, el, evec, SRC, DST, mask,
                                             aw1, ab1, ag, abn, b1p, ib1, ig, ibn,
                                             ib2, B1, B2, node);

  // 5. node -> bf16
  node_to_bf_k<<<(Nn * DNODE + 255) / 256, 256, 0, stream>>>(node, nodebf);

  // 6. mrg block -> edge output
  gemm_edge_k<<<TILES / 4, 128, 0, stream>>>(nodebf, el, SRC, DST,
                                             mb1, mg, mbn, mb2, B3, B4, edge);
}
